// MultiHeadAttention_38603166056992
// MI455X (gfx1250) — matrix operand order repaired and checked by reading, not compiled, not run on hardware
//
#include <hip/hip_runtime.h>

// ---------- CDNA5 WMMA types ----------
typedef __attribute__((ext_vector_type(16))) __bf16 v16bf;
typedef __attribute__((ext_vector_type(8)))  float  v8f;

union FragBF { v16bf v; unsigned short u[16]; uint4 q[2]; };

__device__ __forceinline__ unsigned short f2bf(float f) {
    unsigned int x = __float_as_uint(f);
    unsigned int r = x + 0x7FFFu + ((x >> 16) & 1u);   // round-to-nearest-even
    return (unsigned short)(r >> 16);
}
__device__ __forceinline__ unsigned int f2bf2(float lo, float hi) {
    return (unsigned int)f2bf(lo) | ((unsigned int)f2bf(hi) << 16);
}

// A fragment, 16x32 bf16, from row-major [M][K] tile: two contiguous 16B runs per lane.
__device__ __forceinline__ v16bf fragA(const unsigned short* t, int ld, int row, int k0, int khalf) {
    FragBF f;
    const unsigned short* p = t + row * ld + k0 + khalf * 8;
    f.q[0] = *(const uint4*)(p);        // K = khalf*8 + 0..7
    f.q[1] = *(const uint4*)(p + 16);   // K = 16 + khalf*8 + 0..7
    return f.v;
}
// B fragment, 32x16 bf16, from a tile stored [N][K]: two contiguous 16B runs per lane.
__device__ __forceinline__ v16bf fragB(const unsigned short* t, int ld, int n, int k0, int khalf) {
    FragBF f;
    const unsigned short* p = t + n * ld + k0 + khalf * 8;
    f.q[0] = *(const uint4*)(p);
    f.q[1] = *(const uint4*)(p + 16);
    return f.v;
}

__device__ __forceinline__ v8f wmma_bf16(v16bf a, v16bf b, v8f c) {
    return __builtin_amdgcn_wmma_f32_16x16x32_bf16(false, a, false, b, (short)0, c, false, false);
}

// Async global -> LDS 128-bit copy (GLOBAL_LOAD_ASYNC_TO_LDS_B128, ASYNCcnt).
// GVS addressing: 64-bit SGPR base + 32-bit per-lane byte offset; VDST = LDS byte addr.
__device__ __forceinline__ void async_g2l_b128(const void* lds, unsigned goff, const void* gbase) {
    asm volatile("global_load_async_to_lds_b128 %0, %1, %2"
                 :: "v"((unsigned)(size_t)lds), "v"(goff), "s"(gbase) : "memory");
}
__device__ __forceinline__ void wait_async0() {
    asm volatile("s_wait_asynccnt 0" ::: "memory");
}

#define D_MODEL 768
#define N3      2304
#define TSEQ    1024
#define NHEAD   12
#define HEADSZ  6291456  /* 8*12*1024*64 elements per K/V output */
#define NKSTEP  24       /* 768/32 */

// ============================================================================
// Kernel 1: QKV = X @ W_attn + b_attn.  Double-buffered LDS, bf16 WMMA.
//   Q -> ws bf16 [bh][t][d]
//   K -> d_out fp32 + ws bf16 [bh][t][d]
//   V -> d_out fp32 + ws bf16 [bh][d][t]  (transposed for P.V B-operand)
// grid: (36 n-tiles, 128 m-tiles), block 128 (4 waves), tile 64x64, K-step 32.
// ============================================================================
__global__ __launch_bounds__(128) void qkv_gemm_kernel(
    const float* __restrict__ X, const float* __restrict__ W, const float* __restrict__ bias,
    float* __restrict__ outK, float* __restrict__ outV,
    unsigned short* __restrict__ Qbf, unsigned short* __restrict__ Kbf, unsigned short* __restrict__ Vbf)
{
    __shared__ __align__(16) unsigned short Xs[2][64 * 32];   // [m][k]
    __shared__ __align__(16) unsigned short Wt[2][64 * 32];   // [n][k] (transposed)
    const int nt = blockIdx.x, mt = blockIdx.y;
    const int tid = threadIdx.x, lane = tid & 31, w = tid >> 5;
    const int khalf = lane >> 4, nlo = lane & 15;

    v8f zf = {0.f, 0.f, 0.f, 0.f, 0.f, 0.f, 0.f, 0.f};
    v8f acc[4];
#pragma unroll
    for (int s = 0; s < 4; ++s) acc[s] = zf;

    float2 xr[8], wr[8];
    // ---- prologue: stage tile 0 ----
#pragma unroll
    for (int ii = 0; ii < 8; ++ii) {
        int i = tid + ii * 128;
        int r = i >> 4, c2 = (i & 15) << 1;
        xr[ii] = *(const float2*)&X[(mt * 64 + r) * D_MODEL + c2];
        int k = i >> 5, n2 = (i & 31) << 1;
        wr[ii] = *(const float2*)&W[k * N3 + nt * 64 + n2];
    }
#pragma unroll
    for (int ii = 0; ii < 8; ++ii) {
        int i = tid + ii * 128;
        ((unsigned int*)&Xs[0][0])[i] = f2bf2(xr[ii].x, xr[ii].y);
        int k = i >> 5, n2 = (i & 31) << 1;
        Wt[0][n2 * 32 + k]       = f2bf(wr[ii].x);
        Wt[0][(n2 + 1) * 32 + k] = f2bf(wr[ii].y);
    }
    __syncthreads();

    for (int kt = 0; kt < NKSTEP; ++kt) {
        const int cur = kt & 1;
        const bool more = (kt + 1 < NKSTEP);
        if (more) {                     // global loads for next tile, in flight over WMMAs
            const int k0n = (kt + 1) * 32;
#pragma unroll
            for (int ii = 0; ii < 8; ++ii) {
                int i = tid + ii * 128;
                int r = i >> 4, c2 = (i & 15) << 1;
                xr[ii] = *(const float2*)&X[(mt * 64 + r) * D_MODEL + k0n + c2];
                int k = i >> 5, n2 = (i & 31) << 1;
                wr[ii] = *(const float2*)&W[(k0n + k) * N3 + nt * 64 + n2];
            }
        }
        if (kt + 2 < NKSTEP) {          // global_prefetch_b8 of the k+2 slab
            __builtin_prefetch(&X[(mt * 64 + (tid >> 1)) * D_MODEL + (kt + 2) * 32 + (tid & 1) * 16], 0, 0);
            __builtin_prefetch(&W[((kt + 2) * 32 + (tid >> 2)) * N3 + nt * 64 + (tid & 3) * 16], 0, 0);
        }
        v16bf a  = fragA(&Xs[cur][0], 32, (w << 4) + nlo, 0, khalf);
        v16bf b0 = fragB(&Wt[cur][0], 32, nlo,      0, khalf);
        v16bf b1 = fragB(&Wt[cur][0], 32, 16 + nlo, 0, khalf);
        v16bf b2 = fragB(&Wt[cur][0], 32, 32 + nlo, 0, khalf);
        v16bf b3 = fragB(&Wt[cur][0], 32, 48 + nlo, 0, khalf);
        acc[0] = wmma_bf16(a, b0, acc[0]);
        acc[1] = wmma_bf16(a, b1, acc[1]);
        acc[2] = wmma_bf16(a, b2, acc[2]);
        acc[3] = wmma_bf16(a, b3, acc[3]);
        if (more) {
            const int nxt = cur ^ 1;
#pragma unroll
            for (int ii = 0; ii < 8; ++ii) {
                int i = tid + ii * 128;
                ((unsigned int*)&Xs[nxt][0])[i] = f2bf2(xr[ii].x, xr[ii].y);
                int k = i >> 5, n2 = (i & 31) << 1;
                Wt[nxt][n2 * 32 + k]       = f2bf(wr[ii].x);
                Wt[nxt][(n2 + 1) * 32 + k] = f2bf(wr[ii].y);
            }
        }
        __syncthreads();
    }

    // ---- epilogue: uniform section per n-tile (768 % 64 == 0) ----
    const int j0 = nt * 64;
    const int sect = j0 / 768;                 // 0=Q, 1=K, 2=V (uniform)
    const int jj0 = j0 - sect * 768;
    float bj[4]; int hh[4], dd[4];
#pragma unroll
    for (int s = 0; s < 4; ++s) {
        int sj = s * 16 + nlo;
        bj[s] = bias[j0 + sj];
        hh[s] = (jj0 + sj) >> 6;
        dd[s] = (jj0 + sj) & 63;
    }
    if (sect == 0) {
#pragma unroll
        for (int r = 0; r < 8; ++r) {
            int m = mt * 64 + w * 16 + r + (khalf << 3);
            int bb = m >> 10, t = m & 1023;
#pragma unroll
            for (int s = 0; s < 4; ++s)
                Qbf[((((bb * NHEAD + hh[s]) << 10) + t) << 6) + dd[s]] = f2bf(acc[s][r] + bj[s]);
        }
    } else if (sect == 1) {
#pragma unroll
        for (int r = 0; r < 8; ++r) {
            int m = mt * 64 + w * 16 + r + (khalf << 3);
            int bb = m >> 10, t = m & 1023;
#pragma unroll
            for (int s = 0; s < 4; ++s) {
                float v = acc[s][r] + bj[s];
                int idx = ((((bb * NHEAD + hh[s]) << 10) + t) << 6) + dd[s];
                outK[idx] = v; Kbf[idx] = f2bf(v);
            }
        }
    } else {
#pragma unroll
        for (int r = 0; r < 8; ++r) {
            int m = mt * 64 + w * 16 + r + (khalf << 3);
            int bb = m >> 10, t = m & 1023;
#pragma unroll
            for (int s = 0; s < 4; ++s) {
                float v = acc[s][r] + bj[s];
                int bh = bb * NHEAD + hh[s];
                outV[((bh << 10) + t) * 64 + dd[s]] = v;
                Vbf[(bh << 16) + (dd[s] << 10) + t] = f2bf(v);   // transposed [bh][d][t]
            }
        }
    }
}

// ============================================================================
// Kernel 2: flash attention, causal, double-buffered K/V tiles staged with
// GLOBAL_LOAD_ASYNC_TO_LDS_B128 (ASYNCcnt) -- no VGPR round-trip.
// grid: (16 q-tiles, 96 b*h), block 128 (4 waves x 16 q rows), k-tiles of 64.
// ============================================================================
__global__ __launch_bounds__(128) void attn_kernel(
    const unsigned short* __restrict__ Qbf, const unsigned short* __restrict__ Kbf,
    const unsigned short* __restrict__ Vbf, unsigned short* __restrict__ Vals)
{
    __shared__ __align__(16) unsigned short Qs[64 * 64];      // [q][d]
    __shared__ __align__(16) unsigned short Ks[2][64 * 64];   // [key][d]
    __shared__ __align__(16) unsigned short Vt[2][64 * 64];   // [d][key]
    __shared__ __align__(16) unsigned short Ps[4][16 * 64];   // per-wave [q][key]

    const int qt = blockIdx.x;
    const int bh = blockIdx.y;
    const int bb = bh / NHEAD, h = bh % NHEAD;
    const unsigned short* Qbase = Qbf + ((size_t)bh << 16);   // [t][d]
    const unsigned short* Kbase = Kbf + ((size_t)bh << 16);   // [t][d]
    const unsigned short* Vbase = Vbf + ((size_t)bh << 16);   // [d][t]

    const int tid = threadIdx.x, lane = tid & 31, w = tid >> 5;
    const int khalf = lane >> 4, nlo = lane & 15;

    // ---- prologue: async-stage Q tile and K/V tile 0 ----
#pragma unroll
    for (int ii = 0; ii < 4; ++ii) {
        int i = tid * 8 + ii * 1024;                          // ushort index, 16B granular
        async_g2l_b128(&Qs[i],    (unsigned)(2u * (qt * 4096 + i)), Qbase);
        async_g2l_b128(&Ks[0][i], (unsigned)(2u * i),               Kbase);
        int d = i >> 6, c = i & 63;
        async_g2l_b128(&Vt[0][i], (unsigned)(2u * ((d << 10) + c)), Vbase);
    }
    wait_async0();
    __syncthreads();

    v16bf aQ0 = fragA(Qs, 64, w * 16 + nlo, 0, khalf);
    v16bf aQ1 = fragA(Qs, 64, w * 16 + nlo, 32, khalf);

    v8f zf = {0.f, 0.f, 0.f, 0.f, 0.f, 0.f, 0.f, 0.f};
    v8f o[4];
#pragma unroll
    for (int s = 0; s < 4; ++s) o[s] = zf;
    float rowM[8], rowL[8];
#pragma unroll
    for (int r = 0; r < 8; ++r) { rowM[r] = -3.0e38f; rowL[r] = 0.f; }

    for (int kt = 0; kt <= qt; ++kt) {
        const int cur = kt & 1;
        const bool more = (kt < qt);
        if (more) {                        // async DMA of next K/V tile overlaps this tile's math
            const int nxt = cur ^ 1;
#pragma unroll
            for (int ii = 0; ii < 4; ++ii) {
                int i = tid * 8 + ii * 1024;
                async_g2l_b128(&Ks[nxt][i], (unsigned)(2u * ((kt + 1) * 4096 + i)), Kbase);
                int d = i >> 6, c = i & 63;
                async_g2l_b128(&Vt[nxt][i], (unsigned)(2u * ((d << 10) + (kt + 1) * 64 + c)), Vbase);
            }
        }
        const unsigned short* Kc = &Ks[cur][0];
        const unsigned short* Vc = &Vt[cur][0];

        // S = Q . K^T (contraction over d=64)
        v8f sa[4];
#pragma unroll
        for (int s = 0; s < 4; ++s) {
            v16bf bk0 = fragB(Kc, 64, s * 16 + nlo, 0,  khalf);
            v16bf bk1 = fragB(Kc, 64, s * 16 + nlo, 32, khalf);
            sa[s] = wmma_bf16(aQ0, bk0, zf);
            sa[s] = wmma_bf16(aQ1, bk1, sa[s]);
        }

        const bool diag = (kt == qt);
#pragma unroll
        for (int s = 0; s < 4; ++s)
#pragma unroll
            for (int r = 0; r < 8; ++r) {
                float v = sa[s][r] * 0.125f;               // 1/sqrt(64)
                if (diag) {
                    int key = kt * 64 + s * 16 + nlo;
                    int q   = qt * 64 + w * 16 + r + (khalf << 3);
                    if (key > q) v = -3.0e38f;
                }
                sa[s][r] = v;
            }

        // online softmax: C-layout rows live in 16-lane halves -> xor-reduce 1,2,4,8
#pragma unroll
        for (int r = 0; r < 8; ++r) {
            float m = fmaxf(fmaxf(sa[0][r], sa[1][r]), fmaxf(sa[2][r], sa[3][r]));
            m = fmaxf(m, __shfl_xor(m, 1, 32));
            m = fmaxf(m, __shfl_xor(m, 2, 32));
            m = fmaxf(m, __shfl_xor(m, 4, 32));
            m = fmaxf(m, __shfl_xor(m, 8, 32));
            float newM = fmaxf(rowM[r], m);
            float f = __expf(rowM[r] - newM);
            rowM[r] = newM;
            float ps = 0.f;
#pragma unroll
            for (int s = 0; s < 4; ++s) {
                float p = __expf(sa[s][r] - newM);
                sa[s][r] = p;
                ps += p;
            }
            ps += __shfl_xor(ps, 1, 32);
            ps += __shfl_xor(ps, 2, 32);
            ps += __shfl_xor(ps, 4, 32);
            ps += __shfl_xor(ps, 8, 32);
            rowL[r] = rowL[r] * f + ps;
#pragma unroll
            for (int s = 0; s < 4; ++s) o[s][r] *= f;
        }

        // re-stage P (wave-private LDS region; same-wave LDS ops are in-order -> no barrier)
        unsigned short* P = &Ps[w][0];
#pragma unroll
        for (int r = 0; r < 8; ++r) {
            int m = r + (khalf << 3);
#pragma unroll
            for (int s = 0; s < 4; ++s) P[m * 64 + s * 16 + nlo] = f2bf(sa[s][r]);
        }
        v16bf aP0 = fragA(P, 64, nlo, 0, khalf);
        v16bf aP1 = fragA(P, 64, nlo, 32, khalf);
#pragma unroll
        for (int s = 0; s < 4; ++s) {
            v16bf bv0 = fragB(Vc, 64, s * 16 + nlo, 0,  khalf);
            v16bf bv1 = fragB(Vc, 64, s * 16 + nlo, 32, khalf);
            o[s] = wmma_bf16(aP0, bv0, o[s]);
            o[s] = wmma_bf16(aP1, bv1, o[s]);
        }

        wait_async0();                  // own async DMAs done before signaling the barrier
        __syncthreads();
    }

    // finalize: values laid out [B,T,H*64] bf16 for the projection GEMM
#pragma unroll
    for (int r = 0; r < 8; ++r) {
        float inv = 1.0f / rowL[r];
        int t = qt * 64 + w * 16 + r + (khalf << 3);
#pragma unroll
        for (int s = 0; s < 4; ++s) {
            int dd = s * 16 + nlo;
            Vals[(size_t)(bb * TSEQ + t) * D_MODEL + h * 64 + dd] = f2bf(o[s][r] * inv);
        }
    }
}

// ============================================================================
// Kernel 3: out = Vals @ W_proj + b_proj. Double-buffered. grid: (12, 128).
// ============================================================================
__global__ __launch_bounds__(128) void proj_gemm_kernel(
    const unsigned short* __restrict__ Vals, const float* __restrict__ W,
    const float* __restrict__ bias, float* __restrict__ out)
{
    __shared__ __align__(16) unsigned short Xs[2][64 * 32];   // [m][k]
    __shared__ __align__(16) unsigned short Wt[2][64 * 32];   // [n][k]
    const int nt = blockIdx.x, mt = blockIdx.y;
    const int tid = threadIdx.x, lane = tid & 31, w = tid >> 5;
    const int khalf = lane >> 4, nlo = lane & 15;

    v8f zf = {0.f, 0.f, 0.f, 0.f, 0.f, 0.f, 0.f, 0.f};
    v8f acc[4];
#pragma unroll
    for (int s = 0; s < 4; ++s) acc[s] = zf;

    unsigned int xr[8]; float2 wr[8];
#pragma unroll
    for (int ii = 0; ii < 8; ++ii) {
        int i = tid + ii * 128;
        int r = i >> 4, c = (i & 15) << 1;
        xr[ii] = *(const unsigned int*)&Vals[(mt * 64 + r) * D_MODEL + c];
        int k = i >> 5, n2 = (i & 31) << 1;
        wr[ii] = *(const float2*)&W[k * D_MODEL + nt * 64 + n2];
    }
#pragma unroll
    for (int ii = 0; ii < 8; ++ii) {
        int i = tid + ii * 128;
        ((unsigned int*)&Xs[0][0])[i] = xr[ii];
        int k = i >> 5, n2 = (i & 31) << 1;
        Wt[0][n2 * 32 + k]       = f2bf(wr[ii].x);
        Wt[0][(n2 + 1) * 32 + k] = f2bf(wr[ii].y);
    }
    __syncthreads();

    for (int kt = 0; kt < NKSTEP; ++kt) {
        const int cur = kt & 1;
        const bool more = (kt + 1 < NKSTEP);
        if (more) {
            const int k0n = (kt + 1) * 32;
#pragma unroll
            for (int ii = 0; ii < 8; ++ii) {
                int i = tid + ii * 128;
                int r = i >> 4, c = (i & 15) << 1;
                xr[ii] = *(const unsigned int*)&Vals[(mt * 64 + r) * D_MODEL + k0n + c];
                int k = i >> 5, n2 = (i & 31) << 1;
                wr[ii] = *(const float2*)&W[(k0n + k) * D_MODEL + nt * 64 + n2];
            }
        }
        if (kt + 2 < NKSTEP) {
            __builtin_prefetch(&Vals[(mt * 64 + (tid >> 1)) * D_MODEL + (kt + 2) * 32 + (tid & 1) * 16], 0, 0);
            __builtin_prefetch(&W[((kt + 2) * 32 + (tid >> 2)) * D_MODEL + nt * 64 + (tid & 3) * 16], 0, 0);
        }
        v16bf a  = fragA(&Xs[cur][0], 32, (w << 4) + nlo, 0, khalf);
        v16bf b0 = fragB(&Wt[cur][0], 32, nlo,      0, khalf);
        v16bf b1 = fragB(&Wt[cur][0], 32, 16 + nlo, 0, khalf);
        v16bf b2 = fragB(&Wt[cur][0], 32, 32 + nlo, 0, khalf);
        v16bf b3 = fragB(&Wt[cur][0], 32, 48 + nlo, 0, khalf);
        acc[0] = wmma_bf16(a, b0, acc[0]);
        acc[1] = wmma_bf16(a, b1, acc[1]);
        acc[2] = wmma_bf16(a, b2, acc[2]);
        acc[3] = wmma_bf16(a, b3, acc[3]);
        if (more) {
            const int nxt = cur ^ 1;
#pragma unroll
            for (int ii = 0; ii < 8; ++ii) {
                int i = tid + ii * 128;
                ((unsigned int*)&Xs[nxt][0])[i] = xr[ii];
                int k = i >> 5, n2 = (i & 31) << 1;
                Wt[nxt][n2 * 32 + k]       = f2bf(wr[ii].x);
                Wt[nxt][(n2 + 1) * 32 + k] = f2bf(wr[ii].y);
            }
        }
        __syncthreads();
    }

    float bj[4];
#pragma unroll
    for (int s = 0; s < 4; ++s) bj[s] = bias[nt * 64 + s * 16 + nlo];
#pragma unroll
    for (int r = 0; r < 8; ++r) {
        int m = mt * 64 + w * 16 + r + (khalf << 3);
#pragma unroll
        for (int s = 0; s < 4; ++s)
            out[(size_t)m * D_MODEL + nt * 64 + s * 16 + nlo] = acc[s][r] + bj[s];
    }
}

extern "C" void kernel_launch(void* const* d_in, const int* in_sizes, int n_in,
                              void* d_out, int out_size, void* d_ws, size_t ws_size,
                              hipStream_t stream) {
    const float* X  = (const float*)d_in[0];   // hidden_states [8,1024,768]
    const float* Wa = (const float*)d_in[1];   // W_attn [768,2304]
    const float* ba = (const float*)d_in[2];   // b_attn [2304]
    const float* Wp = (const float*)d_in[3];   // W_proj [768,768]
    const float* bp = (const float*)d_in[4];   // b_proj [768]

    float* out  = (float*)d_out;               // [8,1024,768]
    float* outK = out + HEADSZ;                // [8,12,1024,64]
    float* outV = out + 2 * HEADSZ;            // [8,12,1024,64]

    unsigned short* Qbf  = (unsigned short*)d_ws;      // bf16 scratch, 12.6MB each
    unsigned short* Kbf  = Qbf + HEADSZ;
    unsigned short* Vbf  = Qbf + 2 * (size_t)HEADSZ;   // transposed [bh][d][t]
    unsigned short* Vals = Qbf + 3 * (size_t)HEADSZ;

    dim3 g1(36, 128);
    qkv_gemm_kernel<<<g1, 128, 0, stream>>>(X, Wa, ba, outK, outV, Qbf, Kbf, Vbf);

    dim3 g2(16, 96);
    attn_kernel<<<g2, 128, 0, stream>>>(Qbf, Kbf, Vbf, Vals);

    dim3 g3(12, 128);
    proj_gemm_kernel<<<g3, 128, 0, stream>>>(Vals, Wp, bp, out);
}
